// SparseGeometricInteraction_69217692942831
// MI455X (gfx1250) — compile-verified, matrix operand order checked by reading, not run on hardware
//
#include <hip/hip_runtime.h>
#include <hip/hip_bf16.h>
#include <stdint.h>

// ---------------- problem constants ----------------
#define BB   4
#define TT   2048
#define DD   1024      // D_DENSE
#define NH   8
#define HD   128
#define NHD  1024      // NH*HD == SPARSE
#define WIN  64
#define KK   32        // top-k
#define RANK 32
#define HID  512       // SPARSE/2

typedef __attribute__((ext_vector_type(16))) __bf16 v16bf;
typedef __attribute__((ext_vector_type(8)))  float  v8f;

// fp32 -> bf16 (round to nearest even)
static __device__ __forceinline__ unsigned short f2bf(float f) {
    uint32_t u = __builtin_bit_cast(uint32_t, f);
    u += 0x7FFFu + ((u >> 16) & 1u);
    return (unsigned short)(u >> 16);
}

union BFrag {
    uint4  q4[2];
    v16bf  v;
};

// LDS tile stride (bf16 elems): 32 data + 8 pad -> 80B rows; ds_load_b128
// stays 16B aligned and 20*n mod 64 is distinct for n<16 (bank spread).
#define LSTR 40

// ---- CDNA5 async copy: 16B global -> LDS per lane, tracked by ASYNCcnt ----
static __device__ __forceinline__ void async_copy_b128(void* lds_ptr, const void* gptr) {
    uint32_t lds_off = (uint32_t)(uintptr_t)lds_ptr;          // LDS aperture: addr[31:0] = LDS byte addr
    unsigned long long ga = (unsigned long long)(uintptr_t)gptr;
    asm volatile("global_load_async_to_lds_b128 %0, %1, off"
                 :: "v"(lds_off), "v"(ga) : "memory");
}
static __device__ __forceinline__ void wait_async_all() {
    asm volatile("s_wait_asynccnt 0x0" ::: "memory");
}

// =====================================================================
// Kernel 0a: fp32 -> bf16 straight convert (x)
// =====================================================================
__global__ void __launch_bounds__(256)
conv_bf16_kernel(const float* __restrict__ in, unsigned short* __restrict__ out, int n)
{
    int i = blockIdx.x * 2048 + threadIdx.x;
    #pragma unroll
    for (int r = 0; r < 8; ++r) {
        int idx = i + r * 256;
        if (idx < n) out[idx] = f2bf(in[idx]);
    }
}

// =====================================================================
// Kernel 0b: encoder weight (NH,DD,HD) -> transposed bf16 (NH,HD,DD)
// out[(h*HD+e)*DD + d] = in[(h*DD+d)*HD + e]
// =====================================================================
__global__ void __launch_bounds__(256)
conv_encw_kernel(const float* __restrict__ in, unsigned short* __restrict__ out)
{
    int i = blockIdx.x * 2048 + threadIdx.x;
    #pragma unroll
    for (int r = 0; r < 8; ++r) {
        int o  = i + r * 256;               // < NH*HD*DD = 1M
        int d  = o & (DD - 1);
        int he = o >> 10;
        int h  = he >> 7, e = he & 127;
        out[o] = f2bf(in[((size_t)h * DD + d) * HD + e]);
    }
}

// =====================================================================
// Kernel 0c: decoder (NHD,DD) -> transposed bf16 (DD,NHD):  out[n*NHD? ]
// out[n*DD_k + k] where A-cols n of output GEMM; decoder is (K=NHD, N=DD)
// out[(n)*NHD + k] = in[k*DD + n]
// =====================================================================
__global__ void __launch_bounds__(256)
conv_dect_kernel(const float* __restrict__ in, unsigned short* __restrict__ out)
{
    int i = blockIdx.x * 2048 + threadIdx.x;
    #pragma unroll
    for (int r = 0; r < 8; ++r) {
        int o = i + r * 256;                // < DD*NHD = 1M
        int k = o & (NHD - 1);
        int n = o >> 10;
        out[o] = f2bf(in[(size_t)k * DD + n]);
    }
}

// =====================================================================
// Kernel 1: fused encoder GEMM, async double-buffered.
// q = relu(xb @ WqT^T), v = relu(xb @ WvT^T)
// A: xb bf16 (8192,1024). B: wqt/wvt bf16 stored transposed (NHD, DD).
// Block 256 thr = 8 waves, BM=64, BN=64; wave -> 16x32 for both outputs.
// =====================================================================
__global__ void __launch_bounds__(256)
encoder_gemm_kernel(const unsigned short* __restrict__ xb,
                    const unsigned short* __restrict__ wqt,  // (NHD, DD)
                    const unsigned short* __restrict__ wvt,  // (NHD, DD)
                    float* __restrict__ qout,                // (B*T, NHD)
                    float* __restrict__ vout)
{
    __shared__ __align__(16) unsigned short ldsA [2][64 * LSTR];
    __shared__ __align__(16) unsigned short ldsBq[2][64 * LSTR];
    __shared__ __align__(16) unsigned short ldsBv[2][64 * LSTR];

    const int tid    = threadIdx.x;
    const int lane   = tid & 31;
    const int wid    = tid >> 5;
    const int wave_m = wid >> 1;
    const int wave_n = wid & 1;
    const int nBase  = blockIdx.x * 64;
    const int mBase  = blockIdx.y * 64;

    // per-thread staging coordinates: one b128 chunk per matrix per tile
    const int sr = tid >> 2;         // 0..63  (row for A / n-row for B)
    const int sc = (tid & 3) * 8;    // 0,8,16,24 (bf16 elem offset)

    v8f aq0 = {}, aq1 = {}, av0 = {}, av1 = {};

    const int rowA   = wave_m * 16 + (lane & 15);
    const int offLo  = (lane < 16) ? 0 : 8;
    const int nFrag0 = wave_n * 32 + (lane & 15);
    const int nFrag1 = nFrag0 + 16;

    // prologue: stage tile 0 into buffer 0
    async_copy_b128(&ldsA [0][sr * LSTR + sc], xb  + (size_t)(mBase + sr) * DD + sc);
    async_copy_b128(&ldsBq[0][sr * LSTR + sc], wqt + (size_t)(nBase + sr) * DD + sc);
    async_copy_b128(&ldsBv[0][sr * LSTR + sc], wvt + (size_t)(nBase + sr) * DD + sc);

    for (int it = 0; it < DD / 32; ++it) {
        const int cur = it & 1;
        wait_async_all();
        __syncthreads();

        if (it + 1 < DD / 32) {   // issue next tile while computing this one
            const int k0 = (it + 1) * 32;
            async_copy_b128(&ldsA [cur ^ 1][sr * LSTR + sc], xb  + (size_t)(mBase + sr) * DD + k0 + sc);
            async_copy_b128(&ldsBq[cur ^ 1][sr * LSTR + sc], wqt + (size_t)(nBase + sr) * DD + k0 + sc);
            async_copy_b128(&ldsBv[cur ^ 1][sr * LSTR + sc], wvt + (size_t)(nBase + sr) * DD + k0 + sc);
        }

        BFrag fa, fq0, fq1, fv0, fv1;
        fa.q4[0]  = *(const uint4*)&ldsA [cur][rowA   * LSTR + offLo];
        fa.q4[1]  = *(const uint4*)&ldsA [cur][rowA   * LSTR + offLo + 16];
        fq0.q4[0] = *(const uint4*)&ldsBq[cur][nFrag0 * LSTR + offLo];
        fq0.q4[1] = *(const uint4*)&ldsBq[cur][nFrag0 * LSTR + offLo + 16];
        fq1.q4[0] = *(const uint4*)&ldsBq[cur][nFrag1 * LSTR + offLo];
        fq1.q4[1] = *(const uint4*)&ldsBq[cur][nFrag1 * LSTR + offLo + 16];
        fv0.q4[0] = *(const uint4*)&ldsBv[cur][nFrag0 * LSTR + offLo];
        fv0.q4[1] = *(const uint4*)&ldsBv[cur][nFrag0 * LSTR + offLo + 16];
        fv1.q4[0] = *(const uint4*)&ldsBv[cur][nFrag1 * LSTR + offLo];
        fv1.q4[1] = *(const uint4*)&ldsBv[cur][nFrag1 * LSTR + offLo + 16];

        aq0 = __builtin_amdgcn_wmma_f32_16x16x32_bf16(false, fa.v, false, fq0.v, (short)0, aq0, false, false);
        aq1 = __builtin_amdgcn_wmma_f32_16x16x32_bf16(false, fa.v, false, fq1.v, (short)0, aq1, false, false);
        av0 = __builtin_amdgcn_wmma_f32_16x16x32_bf16(false, fa.v, false, fv0.v, (short)0, av0, false, false);
        av1 = __builtin_amdgcn_wmma_f32_16x16x32_bf16(false, fa.v, false, fv1.v, (short)0, av1, false, false);
    }

    const int colC = lane & 15;
    const int rAdd = (lane < 16) ? 0 : 8;
    #pragma unroll
    for (int r = 0; r < 8; ++r) {
        int gr = mBase + wave_m * 16 + r + rAdd;
        int gc = nBase + wave_n * 32 + colC;
        qout[(size_t)gr * NHD + gc     ] = fmaxf(aq0[r], 0.f);
        qout[(size_t)gr * NHD + gc + 16] = fmaxf(aq1[r], 0.f);
        vout[(size_t)gr * NHD + gc     ] = fmaxf(av0[r], 0.f);
        vout[(size_t)gr * NHD + gc + 16] = fmaxf(av1[r], 0.f);
    }
}

// =====================================================================
// Kernel 2: sig[b][j] = mean_t q + 0.5 * mean_{t%8==0} q
// =====================================================================
__global__ void __launch_bounds__(256)
sig_kernel(const float* __restrict__ q, float* __restrict__ sig)
{
    const int b = blockIdx.x >> 2;
    const int j = (blockIdx.x & 3) * 256 + threadIdx.x;
    float acc = 0.f, acc8 = 0.f;
    const float* base = q + (size_t)b * TT * NHD + j;
    for (int t = 0; t < TT; ++t) {
        float vv = base[(size_t)t * NHD];
        acc += vv;
        if ((t & 7) == 0) acc8 += vv;
    }
    sig[b * NHD + j] = acc * (1.0f / TT) + 0.5f * acc8 * (1.0f / (TT / 8));
}

// =====================================================================
// Kernel 3: meta-network -> metric.  One block per b.
// =====================================================================
__global__ void __launch_bounds__(256)
metanet_kernel(const float* __restrict__ sig,
               const float* __restrict__ w1, const float* __restrict__ b1,
               const float* __restrict__ w2, const float* __restrict__ b2,
               const float* __restrict__ base_metric,
               float* __restrict__ metric)
{
    __shared__ float s_sig[NHD];
    __shared__ float s_hid[HID];
    const int b = blockIdx.x, tid = threadIdx.x;

    #pragma unroll
    for (int i = 0; i < 4; ++i) s_sig[tid + i * 256] = sig[b * NHD + tid + i * 256];
    __syncthreads();

    #pragma unroll
    for (int jj = 0; jj < 2; ++jj) {
        int j = tid + jj * 256;
        float acc = b1[j];
        for (int i = 0; i < NHD; ++i) acc = fmaf(s_sig[i], w1[(size_t)i * HID + j], acc);
        s_hid[j] = 0.5f * acc * (1.0f + erff(acc * 0.70710678118654752f));  // exact gelu
    }
    __syncthreads();

    #pragma unroll
    for (int jj = 0; jj < 4; ++jj) {
        int j = tid + jj * 256;
        float acc = b2[j];
        for (int i = 0; i < HID; ++i) acc = fmaf(s_hid[i], w2[(size_t)i * NHD + j], acc);
        float m = base_metric[j] + 0.1f * acc;
        metric[b * NHD + j] = (m > 20.f) ? m : log1pf(expf(m));  // softplus
    }
}

// =====================================================================
// Kernel 4: Hm[h] = U[h] @ V[h]   (128x32 @ 32x128)
// =====================================================================
__global__ void __launch_bounds__(256)
hebb_kernel(const float* __restrict__ U, const float* __restrict__ V,
            float* __restrict__ Hm)
{
    const int h = blockIdx.x;
    for (int o = threadIdx.x; o < HD * HD; o += 256) {
        int i = o >> 7, j = o & 127;
        float acc = 0.f;
        #pragma unroll
        for (int r = 0; r < RANK; ++r)
            acc = fmaf(U[(h * HD + i) * RANK + r], V[(h * RANK + r) * HD + j], acc);
        Hm[(size_t)h * HD * HD + o] = acc;
    }
}

// =====================================================================
// Kernel 5: windowed metric top-k attention + hebbian. One wave per (b,h,t).
// =====================================================================
__global__ void __launch_bounds__(256)
window_kernel(const float* __restrict__ q,
              const float* __restrict__ v,
              const float* __restrict__ metric,
              const float* __restrict__ Hm,
              unsigned short* __restrict__ os)
{
    __shared__ float lds_q[8][HD];
    const int lane = threadIdx.x & 31;
    const int wid  = threadIdx.x >> 5;
    const int task = blockIdx.x * 8 + wid;       // (b*NH + h)*T + t
    const int t = task & (TT - 1);
    const int bh = task / TT;
    const int h = bh & (NH - 1);
    const int b = bh >> 3;

    const int e0 = lane * 4;
    const float* qrowbase = q + (size_t)b * TT * NHD + h * HD;
    const float* vrowbase = v + (size_t)b * TT * NHD + h * HD;

    float qv[4], mv[4];
    #pragma unroll
    for (int r = 0; r < 4; ++r) {
        qv[r] = qrowbase[(size_t)t * NHD + e0 + r];
        mv[r] = metric[b * NHD + h * HD + e0 + r];
        lds_q[wid][e0 + r] = qv[r];
    }

    // distances for 64 window candidates (lane parks w=lane, w=lane+32)
    float d0 = 0.f, d1 = 0.f;
    for (int w = 0; w < WIN; ++w) {
        int src = t - w; if (src < 0) src = 0;
        const float* kw = qrowbase + (size_t)src * NHD;
        float p = 0.f;
        #pragma unroll
        for (int r = 0; r < 4; ++r) {
            float d = kw[e0 + r] - qv[r];
            p = fmaf(mv[r] * d, d, p);
        }
        #pragma unroll
        for (int off = 16; off > 0; off >>= 1) p += __shfl_xor(p, off, 32);
        float dist = sqrtf(p + 1e-8f);
        if (lane == (w & 31)) { if (w < 32) d0 = dist; else d1 = dist; }
    }

    // iterative wave-argmin: select 32 smallest, fused weighting + v gather
    float ctx[4] = {0.f, 0.f, 0.f, 0.f};
    float denom = 0.f;
    const float FINF = __builtin_inff();
    for (int k = 0; k < KK; ++k) {
        float bv; int bw;
        if (d0 <= d1) { bv = d0; bw = lane; } else { bv = d1; bw = lane + 32; }
        #pragma unroll
        for (int off = 16; off > 0; off >>= 1) {
            float ov = __shfl_xor(bv, off, 32);
            int   ow = __shfl_xor(bw, off, 32);
            if (ov < bv || (ov == bv && ow < bw)) { bv = ov; bw = ow; }
        }
        float wgt = expf(-bv);
        denom += wgt;
        int src = t - bw; if (src < 0) src = 0;
        const float* vr = vrowbase + (size_t)src * NHD;
        #pragma unroll
        for (int r = 0; r < 4; ++r) ctx[r] = fmaf(wgt, vr[e0 + r], ctx[r]);
        if (lane == (bw & 31)) { if (bw < 32) d0 = FINF; else d1 = FINF; }
    }
    float inv = 1.0f / (denom + 1e-8f);
    #pragma unroll
    for (int r = 0; r < 4; ++r) ctx[r] *= inv;

    // hebbian: ctx += gate*0.1/qn * (q @ Hm[h])
    float p = 0.f;
    #pragma unroll
    for (int r = 0; r < 4; ++r) p = fmaf(qv[r], qv[r], p);
    #pragma unroll
    for (int off = 16; off > 0; off >>= 1) p += __shfl_xor(p, off, 32);
    float qn = sqrtf(p);
    float qscale = (qn > 0.2f) ? (0.1f / fmaxf(qn, 1e-12f)) : 0.f;
    if (qscale != 0.f) {
        float hb[4] = {0.f, 0.f, 0.f, 0.f};
        const float* Hh = Hm + (size_t)h * HD * HD;
        for (int i = 0; i < HD; ++i) {
            float qi = lds_q[wid][i];
            const float* Hr = Hh + (size_t)i * HD;
            #pragma unroll
            for (int r = 0; r < 4; ++r) hb[r] = fmaf(qi, Hr[e0 + r], hb[r]);
        }
        #pragma unroll
        for (int r = 0; r < 4; ++r) ctx[r] = fmaf(qscale, hb[r], ctx[r]);
    }

    unsigned short* orow = os + ((size_t)b * TT + t) * NHD + h * HD + e0;
    #pragma unroll
    for (int r = 0; r < 4; ++r) orow[r] = f2bf(qv[r] * ctx[r]);
}

// =====================================================================
// Kernel 6: decoder GEMM, async double-buffered.
// A = out_sparse bf16 (8192,1024), B = dect bf16 stored transposed (DD,NHD).
// =====================================================================
__global__ void __launch_bounds__(256)
decoder_gemm_kernel(const unsigned short* __restrict__ A,
                    const unsigned short* __restrict__ Bt,   // (DD, NHD)
                    float* __restrict__ C)
{
    __shared__ __align__(16) unsigned short ldsA[2][64 * LSTR];
    __shared__ __align__(16) unsigned short ldsB[2][64 * LSTR];

    const int tid    = threadIdx.x;
    const int lane   = tid & 31;
    const int wid    = tid >> 5;
    const int wave_m = wid >> 1;
    const int wave_n = wid & 1;
    const int nBase  = blockIdx.x * 64;
    const int mBase  = blockIdx.y * 64;

    const int sr = tid >> 2;
    const int sc = (tid & 3) * 8;

    v8f a0 = {}, a1 = {};

    const int rowA   = wave_m * 16 + (lane & 15);
    const int offLo  = (lane < 16) ? 0 : 8;
    const int nFrag0 = wave_n * 32 + (lane & 15);
    const int nFrag1 = nFrag0 + 16;

    async_copy_b128(&ldsA[0][sr * LSTR + sc], A  + (size_t)(mBase + sr) * NHD + sc);
    async_copy_b128(&ldsB[0][sr * LSTR + sc], Bt + (size_t)(nBase + sr) * NHD + sc);

    for (int it = 0; it < NHD / 32; ++it) {
        const int cur = it & 1;
        wait_async_all();
        __syncthreads();

        if (it + 1 < NHD / 32) {
            const int k0 = (it + 1) * 32;
            async_copy_b128(&ldsA[cur ^ 1][sr * LSTR + sc], A  + (size_t)(mBase + sr) * NHD + k0 + sc);
            async_copy_b128(&ldsB[cur ^ 1][sr * LSTR + sc], Bt + (size_t)(nBase + sr) * NHD + k0 + sc);
        }

        BFrag fa, fb0, fb1;
        fa.q4[0]  = *(const uint4*)&ldsA[cur][rowA   * LSTR + offLo];
        fa.q4[1]  = *(const uint4*)&ldsA[cur][rowA   * LSTR + offLo + 16];
        fb0.q4[0] = *(const uint4*)&ldsB[cur][nFrag0 * LSTR + offLo];
        fb0.q4[1] = *(const uint4*)&ldsB[cur][nFrag0 * LSTR + offLo + 16];
        fb1.q4[0] = *(const uint4*)&ldsB[cur][nFrag1 * LSTR + offLo];
        fb1.q4[1] = *(const uint4*)&ldsB[cur][nFrag1 * LSTR + offLo + 16];

        a0 = __builtin_amdgcn_wmma_f32_16x16x32_bf16(false, fa.v, false, fb0.v, (short)0, a0, false, false);
        a1 = __builtin_amdgcn_wmma_f32_16x16x32_bf16(false, fa.v, false, fb1.v, (short)0, a1, false, false);
    }

    const int colC = lane & 15;
    const int rAdd = (lane < 16) ? 0 : 8;
    #pragma unroll
    for (int r = 0; r < 8; ++r) {
        int gr = mBase + wave_m * 16 + r + rAdd;
        int gc = nBase + wave_n * 32 + colC;
        C[(size_t)gr * DD + gc     ] = a0[r];
        C[(size_t)gr * DD + gc + 16] = a1[r];
    }
}

// =====================================================================
extern "C" void kernel_launch(void* const* d_in, const int* in_sizes, int n_in,
                              void* d_out, int out_size, void* d_ws, size_t ws_size,
                              hipStream_t stream) {
    const float* x    = (const float*)d_in[0];
    const float* eq   = (const float*)d_in[1];
    const float* ev   = (const float*)d_in[2];
    const float* w1   = (const float*)d_in[3];
    const float* b1   = (const float*)d_in[4];
    const float* w2   = (const float*)d_in[5];
    const float* b2   = (const float*)d_in[6];
    const float* bm   = (const float*)d_in[7];
    const float* dec  = (const float*)d_in[8];
    const float* hU   = (const float*)d_in[9];
    const float* hV   = (const float*)d_in[10];
    float* out = (float*)d_out;

    // ---- workspace layout ----
    char* ws = (char*)d_ws;
    size_t off = 0;
    auto alloc = [&](size_t bytes) { char* p = ws + off; off += (bytes + 255) & ~(size_t)255; return p; };

    const size_t MTOK = (size_t)BB * TT;                    // 8192
    float*          q_ws      = (float*)alloc(MTOK * NHD * 4);     // 32 MB
    float*          v_ws      = (float*)alloc(MTOK * NHD * 4);     // 32 MB
    float*          sig_ws    = (float*)alloc(BB * NHD * 4);
    float*          metric_ws = (float*)alloc(BB * NHD * 4);
    float*          hm_ws     = (float*)alloc((size_t)NH * HD * HD * 4);
    unsigned short* os_ws     = (unsigned short*)alloc(MTOK * NHD * 2);     // 16 MB
    unsigned short* xb_ws     = (unsigned short*)alloc(MTOK * DD * 2);      // 16 MB
    unsigned short* wqt_ws    = (unsigned short*)alloc((size_t)NHD * DD * 2);
    unsigned short* wvt_ws    = (unsigned short*)alloc((size_t)NHD * DD * 2);
    unsigned short* dect_ws   = (unsigned short*)alloc((size_t)DD * NHD * 2);

    // 0) one-shot bf16 conversions / transposes
    conv_bf16_kernel<<<(int)((MTOK * DD) / 2048), 256, 0, stream>>>(x, xb_ws, (int)(MTOK * DD));
    conv_encw_kernel<<<(NHD * DD) / 2048, 256, 0, stream>>>(eq, wqt_ws);
    conv_encw_kernel<<<(NHD * DD) / 2048, 256, 0, stream>>>(ev, wvt_ws);
    conv_dect_kernel<<<(DD * NHD) / 2048, 256, 0, stream>>>(dec, dect_ws);

    // 1) fused encoder GEMMs (WMMA bf16, async double-buffered staging)
    encoder_gemm_kernel<<<dim3(NHD / 64, (int)(MTOK / 64)), 256, 0, stream>>>(
        xb_ws, wqt_ws, wvt_ws, q_ws, v_ws);

    // 2) signature reduction over T
    sig_kernel<<<BB * (NHD / 256), 256, 0, stream>>>(q_ws, sig_ws);

    // 3) meta-network -> metric
    metanet_kernel<<<BB, 256, 0, stream>>>(sig_ws, w1, b1, w2, b2, bm, metric_ws);

    // 4) Hm = U @ V per head
    hebb_kernel<<<NH, 256, 0, stream>>>(hU, hV, hm_ws);

    // 5) windowed top-k metric attention + hebbian, one wave per (b,h,t)
    window_kernel<<<(BB * NH * TT) / 8, 256, 0, stream>>>(
        q_ws, v_ws, metric_ws, hm_ws, os_ws);

    // 6) decoder GEMM (WMMA bf16, async double-buffered staging)
    decoder_gemm_kernel<<<dim3(DD / 64, (int)(MTOK / 64)), 256, 0, stream>>>(
        os_ws, dect_ws, out);
}
